// ODELSTMCell_40458591928886
// MI455X (gfx1250) — matrix operand order repaired and checked by reading, not compiled, not run on hardware
//
#include <hip/hip_runtime.h>
#include <stdint.h>

typedef __attribute__((ext_vector_type(16))) __bf16 v16bf;
typedef __attribute__((ext_vector_type(2)))  __bf16 v2bf;
typedef __attribute__((ext_vector_type(2)))  float  v2f;
typedef __attribute__((ext_vector_type(8)))  float  v8f;

#define B_TOT   65536
#define IN_DIM  64
#define H_DIM   128
#define OH_DIM  64
#define HP      136   // padded f32 row stride (floats) for h/k1/k2 arrays
#define TSTR    72    // staged tanh-activation row stride (bf16 elems)
#define YSTR    136   // staged y row stride (bf16 elems)

// d_ws dword layout (fragment-linear packed bf16 weights + f32 biases)
#define WS_WG    0        // 8 hc * 4 gate * 6 kc frags * 256 dw = 49152
#define WS_W1    49152    // 4 nt * 4 kc * 256 = 4096
#define WS_W2    53248    // 8 nt * 2 kc * 256 = 4096
#define WS_BG    57344    // 512 (b_ih + b_hh, f32 bits)
#define WS_B1    57856    // 64
#define WS_B2    57920    // 128
#define WS_TOTAL 58048

// LDS byte offsets
#define OFF_K1    0         // phase2: k1 [128][HP] f32 (phase1: gates weights)
#define OFF_K2    69632     // phase2: k2 [128][HP] f32
#define OFF_STG   139264    // phase2: per-wave bf16 staging, 4352 B each
#define STG_PER_WAVE 4352
#define OFF_H     196608    // h [128][HP] f32
#define OFF_W1F   266240    // 16384 B
#define OFF_W2F   282624    // 16384 B
#define OFF_BG    299008    // 2048 B
#define OFF_B1    301056    // 256 B
#define OFF_B2    301312    // 512 B
#define OFF_DT    301824    // 512 B
#define LDS_BYTES 302336

#if __has_builtin(__builtin_amdgcn_cvt_pk_bf16_f32)
static __device__ __forceinline__ uint32_t bf16pk(float a, float b) {
  union { v2bf v; uint32_t u; } u;
  u.v = __builtin_amdgcn_cvt_pk_bf16_f32(a, b);   // single V_CVT_PK_BF16_F32
  return u.u;
}
static __device__ __forceinline__ uint16_t bf16r(float x) {
  return (uint16_t)(bf16pk(x, 0.0f) & 0xFFFFu);
}
#else
// Let the backend lower fptrunc f32->bf16 (v_cvt_pk_bf16_f32 when available)
static __device__ __forceinline__ uint32_t bf16pk(float a, float b) {
  union { v2bf v; uint32_t u; } u;
  v2f f = {a, b};
  u.v = __builtin_convertvector(f, v2bf);
  return u.u;
}
static __device__ __forceinline__ uint16_t bf16r(float x) {
  union { __bf16 h; uint16_t u; } u;
  u.h = (__bf16)x;
  return u.u;
}
#endif
static __device__ __forceinline__ float sigm(float x) {
  return 1.0f / (1.0f + __expf(-x));
}
static __device__ __forceinline__ float tanh_fast(float x) {
  return 1.0f - 2.0f / (__expf(2.0f * x) + 1.0f);   // saturates correctly at +/-inf
}
static __device__ __forceinline__ v8f vzero() {
  v8f z = {0.f,0.f,0.f,0.f,0.f,0.f,0.f,0.f};
  return z;
}
static __device__ __forceinline__ v8f wmma_bf16(v16bf a, v16bf b, v8f c) {
  return __builtin_amdgcn_wmma_f32_16x16x32_bf16(false, a, false, b, (short)0, c, false, false);
}
// B fragment (32 lanes x 8 dwords, fragment-linear) from LDS
static __device__ __forceinline__ v16bf ldB(const uint32_t* base, int fragIdx, int lane) {
  union { v16bf v; uint4 q[2]; } u;
  const uint4* p = (const uint4*)(base + fragIdx * 256 + lane * 8);
  u.q[0] = p[0]; u.q[1] = p[1];
  return u.v;
}
// A fragment from f32 memory (global or LDS): rowp = row base, col = kbase + half*8
static __device__ __forceinline__ v16bf ldA_f32(const float* rowp, int col) {
  union { v16bf v; uint32_t d[8]; } u;
  const float4* q0 = (const float4*)(rowp + col);
  const float4* q1 = (const float4*)(rowp + col + 16);
  float4 a = q0[0], b = q0[1], c = q1[0], d = q1[1];
  u.d[0] = bf16pk(a.x, a.y); u.d[1] = bf16pk(a.z, a.w);
  u.d[2] = bf16pk(b.x, b.y); u.d[3] = bf16pk(b.z, b.w);
  u.d[4] = bf16pk(c.x, c.y); u.d[5] = bf16pk(c.z, c.w);
  u.d[6] = bf16pk(d.x, d.y); u.d[7] = bf16pk(d.z, d.w);
  return u.v;
}
// A fragment from bf16 LDS staging
static __device__ __forceinline__ v16bf ldA_bf16(const uint16_t* rowp, int col) {
  union { v16bf v; uint4 q[2]; } u;
  u.q[0] = *(const uint4*)(rowp + col);
  u.q[1] = *(const uint4*)(rowp + col + 16);
  return u.v;
}

// ---------------- pre-pass: pack weights into WMMA fragment layout ----------
__global__ void pack_params(const float* __restrict__ W_ih, const float* __restrict__ b_ih,
                            const float* __restrict__ W_hh, const float* __restrict__ b_hh,
                            const float* __restrict__ W1,   const float* __restrict__ b1,
                            const float* __restrict__ W2,   const float* __restrict__ b2,
                            uint32_t* __restrict__ ws) {
  int idx = blockIdx.x * blockDim.x + threadIdx.x;
  if (idx >= WS_TOTAL) return;
  if (idx < WS_W1) {                       // gates weights: B[k][n] = Wcat[n][k], bf16 pairs
    int frag = idx >> 8, lane = (idx >> 3) & 31, v = idx & 7;
    int kc = frag % 6, g = (frag / 6) & 3, hc = frag / 24;
    int nn = lane & 15, hf = lane >> 4;
    int K = kc * 32 + hf * 8 + 2 * v + (v >> 2) * 8;
    int grow = g * 128 + hc * 16 + nn;     // row of W_ih/W_hh (gate output index)
    float f0, f1;
    if (K < 64) { f0 = W_ih[grow * 64 + K];        f1 = W_ih[grow * 64 + K + 1]; }
    else        { f0 = W_hh[grow * 128 + K - 64];  f1 = W_hh[grow * 128 + K - 63]; }
    ws[idx] = bf16pk(f0, f1);
  } else if (idx < WS_W2) {                // W1: (64,128), B[k][n] = W1[n][k]
    int j = idx - WS_W1;
    int frag = j >> 8, lane = (j >> 3) & 31, v = j & 7;
    int kc = frag & 3, nt = frag >> 2;
    int nn = lane & 15, hf = lane >> 4;
    int K = kc * 32 + hf * 8 + 2 * v + (v >> 2) * 8;
    int row = nt * 16 + nn;
    ws[idx] = bf16pk(W1[row * 128 + K], W1[row * 128 + K + 1]);
  } else if (idx < WS_BG) {                // W2: (128,64), B[k][n] = W2[n][k]
    int j = idx - WS_W2;
    int frag = j >> 8, lane = (j >> 3) & 31, v = j & 7;
    int kc = frag & 1, nt = frag >> 1;
    int nn = lane & 15, hf = lane >> 4;
    int K = kc * 32 + hf * 8 + 2 * v + (v >> 2) * 8;
    int row = nt * 16 + nn;
    ws[idx] = bf16pk(W2[row * 64 + K], W2[row * 64 + K + 1]);
  } else if (idx < WS_B1) {
    int j = idx - WS_BG;
    ws[idx] = __float_as_uint(b_ih[j] + b_hh[j]);
  } else if (idx < WS_B2) {
    ws[idx] = __float_as_uint(b1[idx - WS_B1]);
  } else {
    ws[idx] = __float_as_uint(b2[idx - WS_B2]);
  }
}

// ---------------- fused LSTM-cell + RK ODE kernel ---------------------------
__global__ __launch_bounds__(256)
void odelstm_fused(const float* __restrict__ input, const float* __restrict__ h0,
                   const float* __restrict__ c0, const float* __restrict__ ts,
                   const uint32_t* __restrict__ ws, float* __restrict__ out) {
  extern __shared__ __align__(16) char smem[];
  const int tid  = threadIdx.x;
  const int lane = tid & 31;
  const int w    = tid >> 5;
  const int n    = lane & 15;
  const int hf   = lane >> 4;

  // ---- stage packed weights into LDS ----
  {
    const uint4* src = (const uint4*)ws;
    uint4* dstWG = (uint4*)(smem + OFF_K1);            // gates weights live in reuse region
    for (int i = tid; i < 12288; i += 256) dstWG[i] = src[i];     // 49152 dwords
    const uint32_t* src2 = ws + WS_W1;
    uint32_t* dst2 = (uint32_t*)(smem + OFF_W1F);      // W1F,W2F,BG,B1,B2 contiguous
    for (int i = tid; i < 8896; i += 256) dst2[i] = src2[i];
    if (tid < 128) ((float*)(smem + OFF_DT))[tid] = ts[blockIdx.x * 128 + tid];
  }
  __syncthreads();

  const int rowg = blockIdx.x * 128 + w * 16;          // first global row of this wave
  float* hl = (float*)(smem + OFF_H);

  // ---- phase 1: gates GEMM (K=192) + LSTM nonlinearity ----
  {
    v16bf A[6];
    const float* ip = input + (size_t)(rowg + n) * IN_DIM;
    const float* hp = h0 + (size_t)(rowg + n) * H_DIM;
    A[0] = ldA_f32(ip, 0 + hf * 8);
    A[1] = ldA_f32(ip, 32 + hf * 8);
#pragma unroll
    for (int kc = 0; kc < 4; ++kc) A[2 + kc] = ldA_f32(hp, kc * 32 + hf * 8);

    const uint32_t* wg  = (const uint32_t*)(smem + OFF_K1);
    const float*    bgp = (const float*)(smem + OFF_BG);
    float* outc = out + (size_t)B_TOT * H_DIM;

    for (int hc = 0; hc < 8; ++hc) {
      v8f ai = vzero(), af = vzero(), ag = vzero(), ao = vzero();
#pragma unroll
      for (int kc = 0; kc < 6; ++kc) {
        int fb = hc * 24 + kc;
        ai = wmma_bf16(A[kc], ldB(wg, fb + 0,  lane), ai);
        af = wmma_bf16(A[kc], ldB(wg, fb + 6,  lane), af);
        ag = wmma_bf16(A[kc], ldB(wg, fb + 12, lane), ag);
        ao = wmma_bf16(A[kc], ldB(wg, fb + 18, lane), ao);
      }
      int col = hc * 16 + n;
      float bi = bgp[col], bf_ = bgp[128 + col], bg_ = bgp[256 + col], bo = bgp[384 + col];
#pragma unroll
      for (int r = 0; r < 8; ++r) {
        int growr = rowg + r + 8 * hf;
        float c0v = __builtin_nontemporal_load(&c0[(size_t)growr * H_DIM + col]);
        float ig = sigm(ai[r] + bi);
        float fg = sigm(af[r] + bf_);
        float gg = tanh_fast(ag[r] + bg_);
        float og = sigm(ao[r] + bo);
        float cv = fg * c0v + ig * gg;
        float hv = og * tanh_fast(cv);
        __builtin_nontemporal_store(cv, &outc[(size_t)growr * H_DIM + col]);  // c output
        hl[(w * 16 + r + 8 * hf) * HP + col] = hv;                            // h -> LDS
      }
    }
  }
  __syncthreads();   // all waves done with gates weights; reuse region becomes k1/k2/staging

  // ---- phase 2: RK integration, each f_ode = two WMMA GEMMs ----
  float* k1l = (float*)(smem + OFF_K1);
  float* k2l = (float*)(smem + OFF_K2);
  uint16_t* stg16 = (uint16_t*)(smem + OFF_STG + w * STG_PER_WAVE);
  const uint32_t* w1f = (const uint32_t*)(smem + OFF_W1F);
  const uint32_t* w2f = (const uint32_t*)(smem + OFF_W2F);
  const float* b1p = (const float*)(smem + OFF_B1);
  const float* b2p = (const float*)(smem + OFF_B2);

  float dtr[8];
#pragma unroll
  for (int r = 0; r < 8; ++r) dtr[r] = ((float*)(smem + OFF_DT))[w * 16 + r + 8 * hf];

  auto gemm_ode = [&](v16bf* Afr, v8f* kk) {
    v8f t[4];
#pragma unroll
    for (int nt = 0; nt < 4; ++nt) t[nt] = vzero();
#pragma unroll
    for (int kc = 0; kc < 4; ++kc)
#pragma unroll
      for (int nt = 0; nt < 4; ++nt)
        t[nt] = wmma_bf16(Afr[kc], ldB(w1f, nt * 4 + kc, lane), t[nt]);
#pragma unroll
    for (int nt = 0; nt < 4; ++nt) {
      float bb = b1p[nt * 16 + n];
#pragma unroll
      for (int r = 0; r < 8; ++r)
        stg16[(r + 8 * hf) * TSTR + nt * 16 + n] = bf16r(tanh_fast(t[nt][r] + bb));
    }
    asm volatile("s_wait_dscnt 0" ::: "memory");
    v16bf Tf[2];
#pragma unroll
    for (int kc = 0; kc < 2; ++kc) Tf[kc] = ldA_bf16(stg16 + n * TSTR, kc * 32 + hf * 8);
#pragma unroll
    for (int nt = 0; nt < 8; ++nt) kk[nt] = vzero();
#pragma unroll
    for (int kc = 0; kc < 2; ++kc)
#pragma unroll
      for (int nt = 0; nt < 8; ++nt)
        kk[nt] = wmma_bf16(Tf[kc], ldB(w2f, nt * 2 + kc, lane), kk[nt]);
#pragma unroll
    for (int nt = 0; nt < 8; ++nt) {
      float bb = b2p[nt * 16 + n];
#pragma unroll
      for (int r = 0; r < 8; ++r) kk[nt][r] += bb;
    }
  };

  const float THIRD = 1.0f / 3.0f;
  v16bf Afr[4];
  v8f kk[8];

  // k1 = f(h);  y2 = h + dt*k1/3
#pragma unroll
  for (int kc = 0; kc < 4; ++kc)
    Afr[kc] = ldA_f32(hl + (size_t)(w * 16 + n) * HP, kc * 32 + hf * 8);
  gemm_ode(Afr, kk);
#pragma unroll
  for (int nt = 0; nt < 8; ++nt) {
    int col = nt * 16 + n;
#pragma unroll
    for (int r = 0; r < 8; ++r) {
      int lr = (w * 16 + r + 8 * hf) * HP + col;
      float hv = hl[lr], k1v = kk[nt][r];
      k1l[lr] = k1v;
      stg16[(r + 8 * hf) * YSTR + col] = bf16r(hv + dtr[r] * (k1v * THIRD));
    }
  }
  asm volatile("s_wait_dscnt 0" ::: "memory");

  // k2 = f(y2);  y3 = h + dt*(k2 - k1/3)
#pragma unroll
  for (int kc = 0; kc < 4; ++kc) Afr[kc] = ldA_bf16(stg16 + n * YSTR, kc * 32 + hf * 8);
  gemm_ode(Afr, kk);
#pragma unroll
  for (int nt = 0; nt < 8; ++nt) {
    int col = nt * 16 + n;
#pragma unroll
    for (int r = 0; r < 8; ++r) {
      int lr = (w * 16 + r + 8 * hf) * HP + col;
      float hv = hl[lr], k1v = k1l[lr], k2v = kk[nt][r];
      k2l[lr] = k2v;
      stg16[(r + 8 * hf) * YSTR + col] = bf16r(hv + dtr[r] * (k2v - k1v * THIRD));
    }
  }
  asm volatile("s_wait_dscnt 0" ::: "memory");

  // k3 = f(y3);  y4 = h + dt*(k1 - k2 + k3);  k1l <- k1 + 3k2 + 3k3
#pragma unroll
  for (int kc = 0; kc < 4; ++kc) Afr[kc] = ldA_bf16(stg16 + n * YSTR, kc * 32 + hf * 8);
  gemm_ode(Afr, kk);
#pragma unroll
  for (int nt = 0; nt < 8; ++nt) {
    int col = nt * 16 + n;
#pragma unroll
    for (int r = 0; r < 8; ++r) {
      int lr = (w * 16 + r + 8 * hf) * HP + col;
      float hv = hl[lr], k1v = k1l[lr], k2v = k2l[lr], k3v = kk[nt][r];
      stg16[(r + 8 * hf) * YSTR + col] = bf16r(hv + dtr[r] * (k1v - k2v + k3v));
      k1l[lr] = k1v + 3.0f * (k2v + k3v);
    }
  }
  asm volatile("s_wait_dscnt 0" ::: "memory");

  // k4 = f(y4);  h_out = h + dt*(S + k4)/8
#pragma unroll
  for (int kc = 0; kc < 4; ++kc) Afr[kc] = ldA_bf16(stg16 + n * YSTR, kc * 32 + hf * 8);
  gemm_ode(Afr, kk);
#pragma unroll
  for (int nt = 0; nt < 8; ++nt) {
    int col = nt * 16 + n;
#pragma unroll
    for (int r = 0; r < 8; ++r) {
      int lr = (w * 16 + r + 8 * hf) * HP + col;
      float hv = hl[lr], sp = k1l[lr];
      float ho = hv + dtr[r] * (sp + kk[nt][r]) * 0.125f;
      __builtin_nontemporal_store(ho, &out[(size_t)(rowg + r + 8 * hf) * H_DIM + col]);
    }
  }
}

extern "C" void kernel_launch(void* const* d_in, const int* in_sizes, int n_in,
                              void* d_out, int out_size, void* d_ws, size_t ws_size,
                              hipStream_t stream) {
  const float* input = (const float*)d_in[0];
  const float* h0    = (const float*)d_in[1];
  const float* c0    = (const float*)d_in[2];
  const float* ts    = (const float*)d_in[3];
  const float* W_ih  = (const float*)d_in[4];
  const float* b_ih  = (const float*)d_in[5];
  const float* W_hh  = (const float*)d_in[6];
  const float* b_hh  = (const float*)d_in[7];
  const float* W1    = (const float*)d_in[8];
  const float* b1    = (const float*)d_in[9];
  const float* W2    = (const float*)d_in[10];
  const float* b2    = (const float*)d_in[11];
  float* out = (float*)d_out;
  uint32_t* ws = (uint32_t*)d_ws;

  pack_params<<<(WS_TOTAL + 255) / 256, 256, 0, stream>>>(W_ih, b_ih, W_hh, b_hh,
                                                          W1, b1, W2, b2, ws);
  odelstm_fused<<<B_TOT / 128, 256, LDS_BYTES, stream>>>(input, h0, c0, ts, ws, out);
}
